// QKRandomShuffleAttention_32452772889005
// MI455X (gfx1250) — matrix operand order repaired and checked by reading, not compiled, not run on hardware
//
#include <hip/hip_runtime.h>

// Problem constants (from reference)
constexpr int SEQ = 1024;   // L
constexpr int DM  = 768;    // DIM == INNER
constexpr int NH  = 12;     // HEADS
constexpr int DH  = 64;     // DIM_HEAD
constexpr int NB  = 16;     // B

typedef __attribute__((ext_vector_type(16))) _Float16 v16h;
typedef __attribute__((ext_vector_type(8)))  _Float16 v8h;
typedef __attribute__((ext_vector_type(8)))  float    v8f;
typedef __attribute__((ext_vector_type(4)))  unsigned int v4u;
typedef __attribute__((ext_vector_type(8)))  int      v8i;
typedef __attribute__((ext_vector_type(4)))  int      v4i;

#if __has_builtin(__builtin_amdgcn_tensor_load_to_lds) && \
    __has_builtin(__builtin_amdgcn_s_wait_tensorcnt)
#define USE_TDM 1
#else
#define USE_TDM 0
#endif

// TDM-staged weight slab in LDS: 768 rows x 64 f16 cols, padded to 72-elem rows
constexpr int WROW = 72;                       // 64 data + 8 pad f16 (144 B rows)
constexpr unsigned WLDS_BYTES = (unsigned)DM * WROW * 2;   // 110592 B

union AFrag { v16h v; v8h h[2]; _Float16 e[16]; };

// A-operand (16x32 f16): lane = row M, elements = [8*half..+7] and [16+8*half..+7]
__device__ inline v16h load_a16(const _Float16* __restrict__ row, int colbase) {
  AFrag f;
  f.h[0] = *(const v8h*)(row + colbase);
  f.h[1] = *(const v8h*)(row + colbase + 16);
  return f.v;
}

__device__ inline v8f wmma_f16(v16h a, v16h b, v8f c) {
  return __builtin_amdgcn_wmma_f32_16x16x32_f16(false, a, false, b, (short)0, c, false, false);
}

#if USE_TDM
// DMA a 768x64 f16 tile (row-major, row stride DM elems) from `gsrc` into LDS
// offset `ldsoff`, padding each 128 B row with 16 B (-> 144 B LDS row stride).
// 6-arg builtin variant: (v4u g0, v8i g1, v4i g2, v4i g3, v8i g4, i32 cpol).
__device__ inline void tdm_load_weight_slab(const _Float16* gsrc, unsigned ldsoff) {
  const unsigned long long ga = (unsigned long long)(uintptr_t)gsrc;
  v4u g0 = { 1u,                                   // count=1, user descriptor
             ldsoff,                               // lds_addr
             (unsigned)(ga & 0xffffffffu),
             (unsigned)(((ga >> 32) & 0x01ffffffu) | (2u << 30)) };  // type=2
  v8i g1 = { (int)((1u << 16)       // data_size = 2 bytes
                 | (1u << 20)       // pad_enable
                 | (4u << 22)       // pad_interval: 32 DWORDs (one 64-elem row)
                 | (3u << 25)),     // pad_amount: 4 DWORDs (16 B)
             (int)(768u << 16),     // tensor_dim0 = 768
             (int)(768u << 16),     // tensor_dim1 = 768
             (int)(64u << 16),      // tile_dim0 = 64
             768,                   // tile_dim1 = 768 (tile_dim2 = 0)
             DM,                    // tensor_dim0_stride = 768
             0, 0 };
  v4i z4 = { 0, 0, 0, 0 };
  v8i z8 = { 0, 0, 0, 0, 0, 0, 0, 0 };
  __builtin_amdgcn_tensor_load_to_lds(g0, g1, z4, z4, z8, 0);
  __builtin_amdgcn_s_wait_tensorcnt(0);
}
#endif

// ---------------------------------------------------------------------------
// K1: convert the four 768x768 f32 weight matrices to f16 (row-major [K][N])
// ---------------------------------------------------------------------------
__global__ void cvt_weights(const float* __restrict__ wq, const float* __restrict__ wk,
                            const float* __restrict__ wv, const float* __restrict__ wo,
                            _Float16* __restrict__ dst) {
  size_t i = (size_t)blockIdx.x * blockDim.x + threadIdx.x;
  const size_t N = (size_t)DM * DM;
  if (i < N) {
    dst[i]         = (_Float16)wq[i];
    dst[N + i]     = (_Float16)wk[i];
    dst[2 * N + i] = (_Float16)wv[i];
    dst[3 * N + i] = (_Float16)wo[i];
  }
}

// ---------------------------------------------------------------------------
// K2: fused gather + projection GEMM.
//   mode 0: q = gather(x,permq) @ Wq  -> [b][h][l][dh]
//   mode 1: k = gather(x,permk) @ Wk  -> [b][h][dh][l]   (transposed for QK^T B-operand)
//   mode 2: v = x @ Wv                -> [b][h][l][dh]
// Block = 8 waves = 128 rows x 64 cols; weight stripe staged in LDS via TDM.
// ---------------------------------------------------------------------------
__global__ void proj_kernel(const float* __restrict__ x,
                            const int* __restrict__ permq,
                            const int* __restrict__ permk,
                            const _Float16* __restrict__ Wb,
                            _Float16* __restrict__ qout,
                            _Float16* __restrict__ kTout,
                            _Float16* __restrict__ vout) {
  const int mode = blockIdx.z / NB;
  const int b    = blockIdx.z % NB;
  const int wave = threadIdx.x >> 5;
  const int lane = threadIdx.x & 31;
  const int half = lane >> 4;
  const int lr   = lane & 15;
  const int Mbase = blockIdx.x * 128 + wave * 16;   // row (l) tile base
  const int Nbase = blockIdx.y * 64;                // output-column tile base
  const int row   = Mbase + lr;
  const float* xb = x + (size_t)b * SEQ * DM;
  const int* perm = (mode == 0) ? permq : (mode == 1) ? permk : nullptr;
  const _Float16* W = Wb + (size_t)mode * DM * DM;

#if USE_TDM
  extern __shared__ _Float16 Wlds[];
  if (__builtin_amdgcn_readfirstlane((int)(threadIdx.x >> 5)) == 0)
    tdm_load_weight_slab(W + Nbase, (unsigned)(uintptr_t)&Wlds[0]);
  __syncthreads();
#endif

  v8f acc[4] = {};
  for (int kk = 0; kk < DM / 32; ++kk) {
    AFrag a;
    #pragma unroll
    for (int c = 0; c < 2; ++c) {
      const int colbase = kk * 32 + 8 * half + 16 * c;
      const int i0 = row * DM + colbase;
      if (perm) {
        #pragma unroll
        for (int e = 0; e < 8; ++e) a.e[8 * c + e] = (_Float16)xb[perm[i0 + e]];
      } else {
        #pragma unroll
        for (int e = 0; e < 8; ++e) a.e[8 * c + e] = (_Float16)xb[i0 + e];
      }
    }
    // B-operand: lane holds col n = lr, elements k = kk*32 + 8*half + (s&7) + 16*(s>>3)
    v16h bf[4];
#if USE_TDM
    const _Float16* wrow = Wlds + (size_t)(kk * 32 + 8 * half) * WROW + lr;
    const int wst = WROW;
#else
    const _Float16* wrow = W + (size_t)(kk * 32 + 8 * half) * DM + Nbase + lr;
    const int wst = DM;
#endif
    #pragma unroll
    for (int nt = 0; nt < 4; ++nt)
      #pragma unroll
      for (int s = 0; s < 16; ++s) bf[nt][s] = wrow[(size_t)((s & 7) + 16 * (s >> 3)) * wst + nt * 16];
    #pragma unroll
    for (int nt = 0; nt < 4; ++nt) acc[nt] = wmma_f16(a.v, bf[nt], acc[nt]);
  }
  #pragma unroll
  for (int nt = 0; nt < 4; ++nt) {
    const int nglob = Nbase + nt * 16 + lr;         // C layout: lane = column
    const int h = nglob >> 6, dh = nglob & 63;
    #pragma unroll
    for (int r = 0; r < 8; ++r) {                   // C layout: vgpr r = row 8*half+r
      const int l = Mbase + 8 * half + r;
      const _Float16 val = (_Float16)acc[nt][r];
      if (mode == 1)
        kTout[(((size_t)b * NH + h) * DH + dh) * SEQ + l] = val;
      else if (mode == 0)
        qout[(((size_t)b * NH + h) * SEQ + l) * DH + dh] = val;
      else
        vout[(((size_t)b * NH + h) * SEQ + l) * DH + dh] = val;
    }
  }
}

// ---------------------------------------------------------------------------
// K3: flash attention. One wave owns a 16-query tile for one (b,h); loops over
// keys in tiles of 32 with online softmax. Block = 4 waves = 64 queries.
// ---------------------------------------------------------------------------
__global__ void attn_kernel(const _Float16* __restrict__ q,
                            const _Float16* __restrict__ kT,
                            const _Float16* __restrict__ v,
                            _Float16* __restrict__ aout) {
  __shared__ _Float16 Plds[4][16 * 32];             // per-wave 16x32 P tile (C->A relayout)
  const int wave = threadIdx.x >> 5;
  const int lane = threadIdx.x & 31;
  const int half = lane >> 4;
  const int lr   = lane & 15;
  const int h = blockIdx.y, b = blockIdx.z;
  const int qbase = (blockIdx.x * 4 + wave) * 16;
  const _Float16* qh = q  + ((size_t)b * NH + h) * SEQ * DH;
  const _Float16* kh = kT + ((size_t)b * NH + h) * DH * SEQ;
  const _Float16* vh = v  + ((size_t)b * NH + h) * SEQ * DH;
  _Float16* Pw = &Plds[wave][0];
  const float scale = 0.125f;                       // DH^-0.5

  v16h aq[2];
  #pragma unroll
  for (int kk = 0; kk < 2; ++kk)
    aq[kk] = load_a16(qh + (size_t)(qbase + lr) * DH, kk * 32 + 8 * half);

  float mrow[8], lrow[8];
  #pragma unroll
  for (int r = 0; r < 8; ++r) { mrow[r] = -1e30f; lrow[r] = 0.f; }
  v8f o[4] = {};

  for (int kt = 0; kt < SEQ / 32; ++kt) {
    const int keybase = kt * 32;
    // S = q k^T  (16 queries x 32 keys), contraction over dh in 2 steps of 32
    v8f s0 = {}, s1 = {};
    #pragma unroll
    for (int kk = 0; kk < 2; ++kk) {
      const _Float16* krow = kh + (size_t)(kk * 32 + 8 * half) * SEQ + keybase + lr;
      v16h b0, b1;
      #pragma unroll
      for (int s = 0; s < 16; ++s) {
        b0[s] = krow[(size_t)((s & 7) + 16 * (s >> 3)) * SEQ];
        b1[s] = krow[(size_t)((s & 7) + 16 * (s >> 3)) * SEQ + 16];
      }
      s0 = wmma_f16(aq[kk], b0, s0);
      s1 = wmma_f16(aq[kk], b1, s1);
    }
    if (kt + 1 < SEQ / 32) {  // -> global_prefetch_b8 for next K/V tile
      __builtin_prefetch(kh + (size_t)(16 * half + lr) * SEQ + keybase + 32, 0, 1);
      __builtin_prefetch(vh + (size_t)(keybase + 32 + 16 * half + lr) * DH, 0, 1);
    }
    // online softmax (rows live as vgpr index r = row 8*half+r, cols across 16 lanes)
    float al[8];
    #pragma unroll
    for (int r = 0; r < 8; ++r) {
      float e0 = s0[r] * scale;
      float e1 = s1[r] * scale;
      float t = fmaxf(e0, e1);
      #pragma unroll
      for (int msk = 1; msk < 16; msk <<= 1) t = fmaxf(t, __shfl_xor(t, msk, 32));
      float mn = fmaxf(mrow[r], t);
      float a  = __expf(mrow[r] - mn);
      float p0 = __expf(e0 - mn);
      float p1 = __expf(e1 - mn);
      float rs = p0 + p1;
      #pragma unroll
      for (int msk = 1; msk < 16; msk <<= 1) rs += __shfl_xor(rs, msk, 32);
      mrow[r] = mn;
      lrow[r] = lrow[r] * a + rs;
      al[r]   = a;
      Pw[(8 * half + r) * 32 + lr]      = (_Float16)p0;   // P row-major [query][key]
      Pw[(8 * half + r) * 32 + lr + 16] = (_Float16)p1;
    }
    #pragma unroll
    for (int j = 0; j < 4; ++j)
      #pragma unroll
      for (int r = 0; r < 8; ++r) o[j][r] = o[j][r] * al[r];

    asm volatile("s_wait_dscnt 0" ::: "memory");    // all lanes' P stores visible
    AFrag ap;                                       // P as A-operand
    ap.h[0] = *(const v8h*)(Pw + lr * 32 + 8 * half);
    ap.h[1] = *(const v8h*)(Pw + lr * 32 + 16 + 8 * half);
    #pragma unroll
    for (int j = 0; j < 4; ++j) {                   // O += P @ V, dh in 4 tiles of 16
      v16h bv;
      #pragma unroll
      for (int s = 0; s < 16; ++s)
        bv[s] = vh[(size_t)(keybase + 8 * half + (s & 7) + 16 * (s >> 3)) * DH + j * 16 + lr];
      o[j] = wmma_f16(ap.v, bv, o[j]);
    }
    asm volatile("" ::: "memory");                  // keep next-iter stores after loads
  }
  #pragma unroll
  for (int r = 0; r < 8; ++r) {
    const float inv = 1.0f / lrow[r];
    const int l = qbase + 8 * half + r;
    #pragma unroll
    for (int j = 0; j < 4; ++j) {
      const int col = h * DH + j * 16 + lr;
      aout[((size_t)b * SEQ + l) * DM + col] = (_Float16)(o[j][r] * inv);
    }
  }
}

// ---------------------------------------------------------------------------
// K4: output projection: out(f32, 16384x768) = attn(f16) @ Wo(f16) + bo
// ---------------------------------------------------------------------------
__global__ void outproj_kernel(const _Float16* __restrict__ A,
                               const _Float16* __restrict__ W,
                               const float* __restrict__ bo,
                               float* __restrict__ out) {
  const int wave = threadIdx.x >> 5;
  const int lane = threadIdx.x & 31;
  const int half = lane >> 4;
  const int lr   = lane & 15;
  const int Mbase = blockIdx.x * 128 + wave * 16;
  const int Nbase = blockIdx.y * 64;
  const int row   = Mbase + lr;

#if USE_TDM
  extern __shared__ _Float16 Wlds[];
  if (__builtin_amdgcn_readfirstlane((int)(threadIdx.x >> 5)) == 0)
    tdm_load_weight_slab(W + Nbase, (unsigned)(uintptr_t)&Wlds[0]);
  __syncthreads();
#endif

  v8f acc[4] = {};
  for (int kk = 0; kk < DM / 32; ++kk) {
    v16h a = load_a16(A + (size_t)row * DM, kk * 32 + 8 * half);
    v16h bf[4];
#if USE_TDM
    const _Float16* wrow = Wlds + (size_t)(kk * 32 + 8 * half) * WROW + lr;
    const int wst = WROW;
#else
    const _Float16* wrow = W + (size_t)(kk * 32 + 8 * half) * DM + Nbase + lr;
    const int wst = DM;
#endif
    #pragma unroll
    for (int nt = 0; nt < 4; ++nt)
      #pragma unroll
      for (int s = 0; s < 16; ++s) bf[nt][s] = wrow[(size_t)((s & 7) + 16 * (s >> 3)) * wst + nt * 16];
    #pragma unroll
    for (int nt = 0; nt < 4; ++nt) acc[nt] = wmma_f16(a, bf[nt], acc[nt]);
  }
  #pragma unroll
  for (int nt = 0; nt < 4; ++nt) {
    const int n = Nbase + nt * 16 + lr;
    const float bias = bo[n];
    #pragma unroll
    for (int r = 0; r < 8; ++r) {
      const int m = Mbase + 8 * half + r;
      out[(size_t)m * DM + n] = acc[nt][r] + bias;
    }
  }
}

// ---------------------------------------------------------------------------
// Launch. Workspace (f16 elems): Wb 4*768^2 | q,kT,v 16*12*1024*64 each |
// attn 16*1024*768.  Total ~100.5 MB.
// ---------------------------------------------------------------------------
extern "C" void kernel_launch(void* const* d_in, const int* in_sizes, int n_in,
                              void* d_out, int out_size, void* d_ws, size_t ws_size,
                              hipStream_t stream) {
  const float* x   = (const float*)d_in[0];
  const float* Wq  = (const float*)d_in[1];
  const float* Wk  = (const float*)d_in[2];
  const float* Wv  = (const float*)d_in[3];
  const float* Wo  = (const float*)d_in[4];
  const float* bo  = (const float*)d_in[5];
  const int* permq = (const int*)d_in[6];
  const int* permk = (const int*)d_in[7];

  const size_t WN  = (size_t)DM * DM;
  const size_t QKV = (size_t)NB * NH * SEQ * DH;
  _Float16* Wb  = (_Float16*)d_ws;
  _Float16* qb  = Wb + 4 * WN;
  _Float16* kTb = qb + QKV;
  _Float16* vb  = kTb + QKV;
  _Float16* ab  = vb + QKV;

  const unsigned shmem = USE_TDM ? WLDS_BYTES : 0;

  cvt_weights<<<(unsigned)((WN + 255) / 256), 256, 0, stream>>>(Wq, Wk, Wv, Wo, Wb);
  proj_kernel<<<dim3(SEQ / 128, DM / 64, 3 * NB), 256, shmem, stream>>>(x, permq, permk, Wb,
                                                                        qb, kTb, vb);
  attn_kernel<<<dim3(SEQ / 64, NH, NB), 128, 0, stream>>>(qb, kTb, vb, ab);
  outproj_kernel<<<dim3(NB * SEQ / 128, DM / 64), 256, shmem, stream>>>(ab, Wb + 3 * WN, bo,
                                                                        (float*)d_out);
  (void)in_sizes; (void)n_in; (void)out_size; (void)ws_size;
}
